// LSTMDecoder_78872779423970
// MI455X (gfx1250) — matrix operand order repaired and checked by reading, not compiled, not run on hardware
//
#include <hip/hip_runtime.h>
#include <hip/hip_bf16.h>
#include <math.h>

// LSTM decoder w/ additive attention for MI455X (gfx1250, wave32).
// B=8 L=128 T=512 V=8000 E=H=D=A=256.

typedef __attribute__((ext_vector_type(2)))  float  v2f;
typedef __attribute__((ext_vector_type(4)))  float  v4f;
typedef __attribute__((ext_vector_type(8)))  float  v8f;
typedef __attribute__((ext_vector_type(16))) __bf16 v16bf;

#define DEV __device__ __forceinline__

DEV float fast_tanh(float x) {
#if __has_builtin(__builtin_amdgcn_tanhf)
  return __builtin_amdgcn_tanhf(x);
#else
  return tanhf(x);
#endif
}
DEV float fast_sigmoid(float x) { return 1.0f / (1.0f + __expf(-x)); }
DEV __bf16 to_bf16(float x) { return (__bf16)x; }

// ---------------------------------------------------------------------------
// f32 GEMM: C[M,N] = A[M,K] @ W[N,K]^T (+bias0+bias1).
// One wave per 16x64 macro-tile (1 A fragment feeds 4 WMMAs), using
// V_WMMA_F32_16X16X4_F32.  Requires N % 64 == 0.
// ---------------------------------------------------------------------------
__global__ void __launch_bounds__(256)
gemm_f32_wmma(const float* __restrict__ A, int lda,
              const float* __restrict__ W,
              float* __restrict__ C, int ldc,
              const float* __restrict__ bias0, const float* __restrict__ bias1,
              int M, int N, int K)
{
  const int wave = threadIdx.x >> 5;
  const int lane = threadIdx.x & 31;
  const int hi   = lane >> 4;          // half-wave select
  const int khalf = 2 * hi;            // K offset within 4-wide step
  int job  = blockIdx.x * 8 + wave;    // job = 16 rows x 64 cols
  int ntn4 = N >> 6;
  int tm = job / ntn4, tn4 = job % ntn4;
  if (tm >= (M >> 4)) return;          // wave-uniform
  const int row = tm * 16 + (lane & 15);
  const float* Arow = A + (size_t)row * lda;
  const float* Wp[4];
#pragma unroll
  for (int j = 0; j < 4; ++j)
    Wp[j] = W + (size_t)(tn4 * 64 + j * 16 + (lane & 15)) * K;
  v8f acc[4] = {{}, {}, {}, {}};
  for (int k0 = 0; k0 < K; k0 += 4) {
    v2f a = *(const v2f*)(Arow + k0 + khalf);       // A[row][k0+khalf +0/1]
#pragma unroll
    for (int j = 0; j < 4; ++j) {
      v2f b = *(const v2f*)(Wp[j] + k0 + khalf);    // B[k][col] = W[col][k]
      acc[j] = __builtin_amdgcn_wmma_f32_16x16x4_f32(false, a, false, b,
                                                     (short)0, acc[j], false, false);
    }
    if ((k0 & 31) == 0 && k0 + 32 < K) {            // prefetch weight stream
#pragma unroll
      for (int j = 0; j < 4; ++j)
        __builtin_prefetch(Wp[j] + k0 + 32, 0, 1);
    }
  }
#pragma unroll
  for (int j = 0; j < 4; ++j) {
    int col = tn4 * 64 + j * 16 + (lane & 15);
    float bv = 0.0f;
    if (bias0) bv += bias0[col];
    if (bias1) bv += bias1[col];
    for (int r = 0; r < 8; ++r)
      C[(size_t)(tm * 16 + r + 8 * hi) * ldc + col] = acc[j][r] + bv;
  }
}

// ---------------------------------------------------------------------------
// bf16 GEMM for logits: C[M,N] = A[M,K]bf16 @ W[N,K]bf16^T + bias, f32 out.
// One wave per 16x64 macro-tile, V_WMMA_F32_16X16X32_BF16 (4 WMMAs per
// A-fragment load).  Requires N % 64 == 0.
// ---------------------------------------------------------------------------
__global__ void __launch_bounds__(256)
gemm_bf16_wmma(const __bf16* __restrict__ A, int lda,
               const __bf16* __restrict__ W, int ldw,
               const float* __restrict__ bias,
               float* __restrict__ C, int ldc,
               int M, int N, int K)
{
  const int wave = threadIdx.x >> 5;
  const int lane = threadIdx.x & 31;
  const int hi   = lane >> 4;
  int job  = blockIdx.x * 8 + wave;
  int ntn4 = N >> 6;
  int tm = job / ntn4, tn4 = job % ntn4;
  if (tm >= (M >> 4)) return;
  const int row = tm * 16 + (lane & 15);
  const __bf16* Arow = A + (size_t)row * lda;
  const __bf16* Wp[4];
#pragma unroll
  for (int j = 0; j < 4; ++j)
    Wp[j] = W + (size_t)(tn4 * 64 + j * 16 + (lane & 15)) * ldw + 16 * hi;
  v8f acc[4] = {{}, {}, {}, {}};
  for (int k0 = 0; k0 < K; k0 += 32) {
    union { v16bf v; v4f f[2]; } a;
    int a0 = k0 + 16 * hi;                          // A: 16 contiguous halves
    a.f[0] = *(const v4f*)(Arow + a0);
    a.f[1] = *(const v4f*)(Arow + a0 + 8);
#pragma unroll
    for (int j = 0; j < 4; ++j) {
      v16bf b = *(const v16bf*)(Wp[j] + k0);        // B: 16 contiguous halves
      acc[j] = __builtin_amdgcn_wmma_f32_16x16x32_bf16(false, a.v, false, b,
                                                       (short)0, acc[j], false, false);
    }
    if (k0 + 32 < K) {                              // prefetch weight stream
#pragma unroll
      for (int j = 0; j < 4; ++j)
        __builtin_prefetch(Wp[j] + k0 + 32, 0, 1);
    }
  }
#pragma unroll
  for (int j = 0; j < 4; ++j) {
    int col = tn4 * 64 + j * 16 + (lane & 15);
    float bv = bias ? bias[col] : 0.0f;
    for (int r = 0; r < 8; ++r)
      C[(size_t)(tm * 16 + r + 8 * hi) * ldc + col] = acc[j][r] + bv;
  }
}

// ---------------------------------------------------------------------------
// LSTM recurrence: 8 WGs, WG wg owns hidden slice [32*wg, 32*wg+32).
// Waves 2g..2g+1 compute gate-type g columns for that slice via bf16 WMMA
// with register-resident w_hh fragments; global spin-barrier per step.
// hbuf = double-buffered h_t (2 x 8 x 256 f32).
// ---------------------------------------------------------------------------
__global__ void __launch_bounds__(256)
lstm_kernel(const float* __restrict__ x_gates,   // [1024][1024] incl. biases
            const __bf16* __restrict__ whh_bf,   // [1024][256]
            float* __restrict__ hbuf,            // [2][8][256]
            float* __restrict__ feat,            // [1024][512], cols 0..255 = h
            int* __restrict__ bar)
{
  __shared__ __bf16 h_bf[16][264];               // padded rows (bank spread)
  __shared__ float  c_s[8][32];
  __shared__ float  g_s[4][8][32];

  const int tid  = threadIdx.x;
  const int wave = tid >> 5;
  const int lane = tid & 31;
  const int hi   = lane >> 4;
  const int wg   = blockIdx.x;                   // 0..7
  const int gt   = wave >> 1;                    // gate type i/f/g/o
  const int sub  = wave & 1;
  const int col  = gt * 256 + wg * 32 + sub * 16 + (lane & 15); // gate row

  // Persistent B fragments: w_hh^T columns for this wave (K=256 -> 8 frags).
  v16bf Bf[8];
  {
    const __bf16* wr = whh_bf + (size_t)col * 256;
#pragma unroll
    for (int kf = 0; kf < 8; ++kf)
      Bf[kf] = *(const v16bf*)(wr + kf * 32 + 16 * hi);
  }

  for (int i = tid; i < 16 * 264; i += 256) ((__bf16*)h_bf)[i] = (__bf16)0.0f;
  ((float*)c_s)[tid & 255] = 0.0f;
  __syncthreads();

  for (int t = 0; t < 128; ++t) {
    // gates = x_gates[b,t,:] + h_{t-1} @ w_hh^T
    v8f acc;
#pragma unroll
    for (int r = 0; r < 8; ++r) {
      int bt = r + 8 * hi;                       // batch (rows 8..15 padded)
      acc[r] = (bt < 8) ? x_gates[(size_t)(bt * 128 + t) * 1024 + col] : 0.0f;
    }
    const __bf16* hr = &h_bf[lane & 15][0];
#pragma unroll
    for (int kf = 0; kf < 8; ++kf) {
      union { v16bf v; v4f f[2]; } a;
      int a0 = kf * 32 + 16 * hi;
      a.f[0] = *(const v4f*)(hr + a0);
      a.f[1] = *(const v4f*)(hr + a0 + 8);
      acc = __builtin_amdgcn_wmma_f32_16x16x32_bf16(false, a.v, false, Bf[kf],
                                                    (short)0, acc, false, false);
    }
#pragma unroll
    for (int r = 0; r < 8; ++r) {
      int bt = r + 8 * hi;
      if (bt < 8) g_s[gt][bt][sub * 16 + (lane & 15)] = acc[r];
    }
    __syncthreads();

    // cell update for (batch, local hidden)
    {
      int bt = tid >> 5, jl = tid & 31;
      float ig = fast_sigmoid(g_s[0][bt][jl]);
      float fg = fast_sigmoid(g_s[1][bt][jl]);
      float gg = fast_tanh(g_s[2][bt][jl]);
      float og = fast_sigmoid(g_s[3][bt][jl]);
      float c  = fg * c_s[bt][jl] + ig * gg;
      float h  = og * fast_tanh(c);
      c_s[bt][jl] = c;
      hbuf[(t & 1) * 2048 + bt * 256 + wg * 32 + jl] = h;
      feat[(size_t)(bt * 128 + t) * 512 + wg * 32 + jl] = h;
    }
    __syncthreads();

    // grid barrier: all 8 WGs finished writing h_t
    if (tid == 0) {
      __threadfence();
      atomicAdd(bar, 1);
      int target = 8 * (t + 1);
      while (__hip_atomic_load(bar, __ATOMIC_ACQUIRE,
                               __HIP_MEMORY_SCOPE_AGENT) < target) { }
    }
    __syncthreads();

    // refresh full h_t (bf16) for next step's A fragments
    for (int i = tid; i < 2048; i += 256)
      h_bf[i >> 8][i & 255] = to_bf16(hbuf[(t & 1) * 2048 + i]);
    __syncthreads();
  }
}

// ---------------------------------------------------------------------------
// Additive attention: e[l,t] = sum_a v[a]*tanh(q[l,a]+k[t,a]); softmax over t;
// ctx = attn @ memory[b] via f32 WMMA.  Block = (b, 16 rows of l).
// ---------------------------------------------------------------------------
__global__ void __launch_bounds__(256)
attn_kernel(const float* __restrict__ q,        // [1024][256]
            const float* __restrict__ kk,       // [4096][256]
            const float* __restrict__ vvec,     // [256]
            const unsigned char* __restrict__ mask, // [8][512]
            const float* __restrict__ memory,   // [8][512][256]
            float* __restrict__ feat)           // [1024][512], cols 256.. = ctx
{
  __shared__ float q_s[16][257];
  __shared__ float k_s[32][257];
  __shared__ float e_s[16][514];
  __shared__ float v_s[256];

  const int tid  = threadIdx.x;
  const int wave = tid >> 5;
  const int lane = tid & 31;
  const int b    = blockIdx.x >> 3;
  const int l0   = (blockIdx.x & 7) * 16;

  v_s[tid & 255] = vvec[tid & 255];
  for (int i = tid; i < 16 * 256; i += 256)
    q_s[i >> 8][i & 255] = q[(size_t)(b * 128 + l0 + (i >> 8)) * 256 + (i & 255)];
  __syncthreads();

  for (int tt = 0; tt < 16; ++tt) {
    const int t0 = tt * 32;
    for (int i = tid; i < 32 * 256; i += 256)
      k_s[i >> 8][i & 255] = kk[(size_t)(b * 512 + t0 + (i >> 8)) * 256 + (i & 255)];
    __syncthreads();
    for (int p = 0; p < 2; ++p) {
      int idx = tid * 2 + p;
      int li = idx >> 5, ti = idx & 31;
      float s = 0.0f;
      for (int a = 0; a < 256; ++a)
        s += v_s[a] * fast_tanh(q_s[li][a] + k_s[ti][a]);
      e_s[li][t0 + ti] = s;
    }
    __syncthreads();
  }

  // mask + softmax over t (wave handles rows 2w, 2w+1)
  for (int rr = 0; rr < 2; ++rr) {
    int li = wave * 2 + rr;
    float mx = -INFINITY;
    for (int t = lane; t < 512; t += 32) {
      float val = mask[b * 512 + t] ? e_s[li][t] : -INFINITY;
      e_s[li][t] = val;
      mx = fmaxf(mx, val);
    }
    for (int off = 16; off > 0; off >>= 1) mx = fmaxf(mx, __shfl_xor(mx, off, 32));
    float sum = 0.0f;
    for (int t = lane; t < 512; t += 32) {
      float ex = __expf(e_s[li][t] - mx);
      e_s[li][t] = ex;
      sum += ex;
    }
    for (int off = 16; off > 0; off >>= 1) sum += __shfl_xor(sum, off, 32);
    float inv = 1.0f / sum;
    for (int t = lane; t < 512; t += 32) e_s[li][t] *= inv;
  }
  __syncthreads();

  // ctx (16x256) = attn (16x512) @ memory[b] (512x256) via f32 WMMA
  const int hi = lane >> 4;
  const int khalf = 2 * hi;
  const float* mem_b = memory + (size_t)b * 512 * 256;
  const float* arow  = &e_s[lane & 15][0];
  for (int nt = wave; nt < 16; nt += 8) {
    const int col = nt * 16 + (lane & 15);
    v8f acc = {};
    for (int k0 = 0; k0 < 512; k0 += 4) {
      v2f a = *(const v2f*)(arow + k0 + khalf);
      v2f bv;
      bv.x = mem_b[(size_t)(k0 + khalf) * 256 + col];
      bv.y = mem_b[(size_t)(k0 + khalf + 1) * 256 + col];
      acc = __builtin_amdgcn_wmma_f32_16x16x4_f32(false, a, false, bv,
                                                  (short)0, acc, false, false);
    }
    for (int r = 0; r < 8; ++r)
      feat[(size_t)(b * 128 + l0 + r + 8 * hi) * 512 + 256 + col] = acc[r];
  }
}

// ---------------------------------------------------------------------------
__global__ void f32_to_bf16_k(const float* __restrict__ in,
                              __bf16* __restrict__ out, int n) {
  int i = blockIdx.x * 256 + threadIdx.x;
  if (i < n) out[i] = to_bf16(in[i]);
}

__global__ void embed_k(const int* __restrict__ ids,
                        const float* __restrict__ ew,
                        float* __restrict__ emb) {
  int row = blockIdx.x;                  // 0..1023 = b*128+l
  int id  = ids[row];
  emb[(size_t)row * 256 + threadIdx.x] = ew[(size_t)id * 256 + threadIdx.x];
}

__global__ void init_k(int* bar) { if (threadIdx.x == 0) bar[0] = 0; }

// ---------------------------------------------------------------------------
extern "C" void kernel_launch(void* const* d_in, const int* in_sizes, int n_in,
                              void* d_out, int out_size, void* d_ws, size_t ws_size,
                              hipStream_t stream)
{
  (void)in_sizes; (void)n_in; (void)out_size; (void)ws_size;
  const int*           ids     = (const int*)          d_in[0];
  const float*         memory  = (const float*)        d_in[1];
  const unsigned char* mmask   = (const unsigned char*)d_in[2];
  const float*         embed_w = (const float*)        d_in[3];
  const float*         w_ih    = (const float*)        d_in[4];
  const float*         w_hh    = (const float*)        d_in[5];
  const float*         b_ih    = (const float*)        d_in[6];
  const float*         b_hh    = (const float*)        d_in[7];
  const float*         wh      = (const float*)        d_in[8];
  const float*         wm      = (const float*)        d_in[9];
  const float*         vvec    = (const float*)        d_in[10];
  const float*         out_w   = (const float*)        d_in[11];
  const float*         out_b   = (const float*)        d_in[12];
  float* logits = (float*)d_out;

  char* ws = (char*)d_ws;
  float*  emb     = (float*) (ws + 0);          // 1024x256      (1.0 MB)
  float*  x_gates = (float*) (ws + 1048576);    // 1024x1024     (4.0 MB)
  float*  feat    = (float*) (ws + 5242880);    // 1024x512      (2.0 MB)
  float*  q       = (float*) (ws + 7340032);    // 1024x256      (1.0 MB)
  float*  kk      = (float*) (ws + 8388608);    // 4096x256      (4.0 MB)
  float*  hbuf    = (float*) (ws + 12582912);   // 2x8x256       (16 KB)
  int*    bar     = (int*)   (ws + 12599296);
  __bf16* whh_bf  = (__bf16*)(ws + 12599552);   // 1024x256      (0.5 MB)
  __bf16* feat_bf = (__bf16*)(ws + 13123840);   // 1024x512      (1.0 MB)
  __bf16* outw_bf = (__bf16*)(ws + 14172416);   // 8000x512      (8.0 MB)

  init_k<<<1, 32, 0, stream>>>(bar);
  f32_to_bf16_k<<<(1024 * 256) / 256, 256, 0, stream>>>(w_hh, whh_bf, 1024 * 256);
  f32_to_bf16_k<<<(8000 * 512) / 256, 256, 0, stream>>>(out_w, outw_bf, 8000 * 512);
  embed_k<<<1024, 256, 0, stream>>>(ids, embed_w, emb);
  // x_gates = emb @ w_ih^T + (b_ih + b_hh): jobs = 64*(1024/64) = 1024
  gemm_f32_wmma<<<128, 256, 0, stream>>>(emb, 256, w_ih, x_gates, 1024,
                                         b_ih, b_hh, 1024, 1024, 256);
  lstm_kernel<<<8, 256, 0, stream>>>(x_gates, whh_bf, hbuf, feat, bar);
  // q = outputs @ wh^T: jobs = 64*(256/64) = 256
  gemm_f32_wmma<<<32, 256, 0, stream>>>(feat, 512, wh, q, 256,
                                        nullptr, nullptr, 1024, 256, 256);
  // k = memory @ wm^T: jobs = 256*4 = 1024
  gemm_f32_wmma<<<128, 256, 0, stream>>>(memory, 256, wm, kk, 256,
                                         nullptr, nullptr, 4096, 256, 256);
  attn_kernel<<<64, 256, 0, stream>>>(q, kk, vvec, mmask, memory, feat);
  f32_to_bf16_k<<<(1024 * 512) / 256, 256, 0, stream>>>(feat, feat_bf, 1024 * 512);
  // logits: jobs = 64*(8000/64) = 8000 -> 1000 blocks
  gemm_bf16_wmma<<<1000, 256, 0, stream>>>(feat_bf, 512, outw_bf, 512, out_b,
                                           logits, 8000, 1024, 8000, 512);
}
